// BasicBlock_73581379715111
// MI455X (gfx1250) — compile-verified, matrix operand order checked by reading, not run on hardware
//
#include <hip/hip_runtime.h>
#include <hip/hip_bf16.h>
#include <cmath>

// ---------------------------------------------------------------------------
// Mamba block forward for MI455X (gfx1250, wave32, WMMA + async LDS staging).
// B=2, L=2048, D_MODEL=768, D_INNER=1536, D_STATE=16, D_CONV=4, DT_RANK=48
// ---------------------------------------------------------------------------

#define BB 2
#define LL 2048
#define DM 768
#define DI 1536
#define DS 16
#define DC 4
#define DTR 48
#define ROWS (BB * LL)          // 4096
#define NPADX 128               // x_proj output rows padded 80 -> 128

#define TILE_M 128
#define TILE_N 128
#define TILE_K 32

#if defined(__HIP_DEVICE_COMPILE__) && __has_builtin(__builtin_amdgcn_global_load_async_to_lds_b128)
#define USE_ASYNC_LDS 1
#else
#define USE_ASYNC_LDS 0
#endif

typedef __attribute__((ext_vector_type(16))) __bf16 v16bf;
typedef __attribute__((ext_vector_type(8)))  __bf16 v8bf;
typedef __attribute__((ext_vector_type(8)))  float  v8f;

// exact parameter type of the async-LDS builtins (from hipcc diagnostic):
// "int __attribute__((__vector_size__(4 * sizeof(int)))) *"
typedef int async_v4i __attribute__((vector_size(4 * sizeof(int))));
#define AS_GLOBAL __attribute__((address_space(1)))
#define AS_LDS    __attribute__((address_space(3)))

// ---- helpers --------------------------------------------------------------

__device__ __forceinline__ __bf16 f2bf(float f) {
  union { float f; unsigned u; } v; v.f = f;
  unsigned r = v.u + 0x7FFFu + ((v.u >> 16) & 1u);   // round-to-nearest-even
  unsigned short h = (unsigned short)(r >> 16);
  return __builtin_bit_cast(__bf16, h);
}

__device__ __forceinline__ float siluf(float x) {
  return x / (1.0f + __expf(-x));
}

__device__ __forceinline__ float softplusf(float x) {
  return (x > 20.0f) ? x : log1pf(__expf(x));
}

__device__ __forceinline__ void wait_async_zero() {
#if __has_builtin(__builtin_amdgcn_s_wait_asynccnt)
  __builtin_amdgcn_s_wait_asynccnt(0);
#else
  asm volatile("s_wait_asynccnt 0" ::: "memory");
#endif
}

// Fragment load (16 rows x 32 K) from an LDS tile with row stride TILE_K.
// ISA lane layout (CDNA5 ISA, 16-bit A/B 16x32):
//   lanes 0-15 : row = lane,    K 0..7  then K 16..23
//   lanes 16-31: row = lane-16, K 8..15 then K 24..31
__device__ __forceinline__ v16bf frag_lds(const __bf16* s, int row0, int lane) {
  int r    = row0 + (lane & 15);
  int koff = (lane >> 4) << 3;
  const __bf16* p = s + r * TILE_K + koff;
  v8bf lo = *(const v8bf*)(p);
  v8bf hi = *(const v8bf*)(p + 16);
  v16bf f;
#pragma unroll
  for (int i = 0; i < 8; ++i) { f[i] = lo[i]; f[i + 8] = hi[i]; }
  return f;
}

// ---- kernel 1: residual add + layernorm -----------------------------------

__global__ __launch_bounds__(256)
void k_addres_ln(const float* __restrict__ x, const float* __restrict__ res,
                 const float* __restrict__ gamma, const float* __restrict__ beta,
                 float* __restrict__ res_out, float* __restrict__ xn) {
  __shared__ float s1[256], s2[256];
  int row = blockIdx.x;                       // 0..ROWS-1
  const float* xr = x   + (size_t)row * DM;
  const float* rr = res + (size_t)row * DM;
  float v0 = xr[threadIdx.x];
  float v1 = xr[threadIdx.x + 256];
  float v2 = xr[threadIdx.x + 512];
  res_out[(size_t)row * DM + threadIdx.x]       = v0 + rr[threadIdx.x];
  res_out[(size_t)row * DM + threadIdx.x + 256] = v1 + rr[threadIdx.x + 256];
  res_out[(size_t)row * DM + threadIdx.x + 512] = v2 + rr[threadIdx.x + 512];

  s1[threadIdx.x] = v0 + v1 + v2;
  s2[threadIdx.x] = v0 * v0 + v1 * v1 + v2 * v2;
  __syncthreads();
  for (int s = 128; s > 0; s >>= 1) {
    if (threadIdx.x < s) {
      s1[threadIdx.x] += s1[threadIdx.x + s];
      s2[threadIdx.x] += s2[threadIdx.x + s];
    }
    __syncthreads();
  }
  float mean = s1[0] * (1.0f / DM);
  float var  = s2[0] * (1.0f / DM) - mean * mean;
  float rstd = rsqrtf(var + 1e-5f);
  float* xo = xn + (size_t)row * DM;
  xo[threadIdx.x]       = (v0 - mean) * rstd * gamma[threadIdx.x]       + beta[threadIdx.x];
  xo[threadIdx.x + 256] = (v1 - mean) * rstd * gamma[threadIdx.x + 256] + beta[threadIdx.x + 256];
  xo[threadIdx.x + 512] = (v2 - mean) * rstd * gamma[threadIdx.x + 512] + beta[threadIdx.x + 512];
}

// ---- kernel 2: fp32 -> bf16 conversion ------------------------------------

__global__ __launch_bounds__(256)
void k_cvt_bf16(const float* __restrict__ in, __bf16* __restrict__ out, int n) {
  int i = blockIdx.x * 256 + threadIdx.x;
  if (i < n) out[i] = f2bf(in[i]);
}

// x_proj weight: pad (80 x 1536) -> (128 x 1536), zero-filled tail rows
__global__ __launch_bounds__(256)
void k_cvt_pad_xproj(const float* __restrict__ w, __bf16* __restrict__ out) {
  int i = blockIdx.x * 256 + threadIdx.x;
  if (i >= NPADX * DI) return;
  int r = i / DI, c = i - r * DI;
  float v = (r < (DTR + 2 * DS)) ? w[r * DI + c] : 0.0f;
  out[i] = f2bf(v);
}

// ---- kernel 3: LDS-staged NT bf16 WMMA GEMM -------------------------------
// C[M,N] (f32, ldc) = A[M,K] (bf16 rm) * B[N,K]^T (bf16 rm).
// Workgroup (8 waves) computes 128x128; per 32-K step it stages 8KB A + 8KB B
// tiles into LDS (async global->LDS when available), then each wave computes
// a 32x64 sub-tile with 8 WMMAs. M,N multiples of 128, K multiple of 32.

__global__ __launch_bounds__(256)
void k_gemm_nt_bf16(const __bf16* __restrict__ A, const __bf16* __restrict__ Bw,
                    float* __restrict__ C, int M, int N, int K, int ldc) {
  __shared__ alignas(16) __bf16 sA[TILE_M * TILE_K];   // 8 KB
  __shared__ alignas(16) __bf16 sB[TILE_N * TILE_K];   // 8 KB

  const int tid  = threadIdx.x;
  const int lane = tid & 31;
  const int wave = tid >> 5;

  const int nblk = N / TILE_N;
  const int m0 = (blockIdx.x / nblk) * TILE_M;
  const int n0 = (blockIdx.x % nblk) * TILE_N;

  const int mw = (wave >> 1) * 32;      // wave's M offset within block tile
  const int nw = (wave & 1) * 64;       // wave's N offset within block tile

  v8f acc[2][4];
#pragma unroll
  for (int i = 0; i < 2; ++i)
#pragma unroll
    for (int j = 0; j < 4; ++j) acc[i][j] = (v8f){};

  for (int k0 = 0; k0 < K; k0 += TILE_K) {
    __syncthreads();                    // protect LDS from previous iteration
    // ---- stage 128x32 A tile and 128x32 B tile: 512 16B chunks each -------
#pragma unroll
    for (int c = tid; c < 512; c += 256) {
      int row  = c >> 2;
      int part = (c & 3) << 3;          // 0,8,16,24 (bf16 elements)
      const __bf16* ga = A  + (size_t)(m0 + row) * K + k0 + part;
      const __bf16* gb = Bw + (size_t)(n0 + row) * K + k0 + part;
      __bf16* la = &sA[row * TILE_K + part];
      __bf16* lb = &sB[row * TILE_K + part];
#if USE_ASYNC_LDS
      __builtin_amdgcn_global_load_async_to_lds_b128(
          (AS_GLOBAL async_v4i*)ga, (AS_LDS async_v4i*)la, 0, 0);
      __builtin_amdgcn_global_load_async_to_lds_b128(
          (AS_GLOBAL async_v4i*)gb, (AS_LDS async_v4i*)lb, 0, 0);
#else
      *(v8bf*)la = *(const v8bf*)ga;
      *(v8bf*)lb = *(const v8bf*)gb;
#endif
    }
#if USE_ASYNC_LDS
    wait_async_zero();                  // this wave's async copies landed
#endif
    __syncthreads();                    // all waves' copies visible

    // ---- compute: 2 A-frags x 4 B-frags -> 8 WMMAs ------------------------
    v16bf a0 = frag_lds(sA, mw,      lane);
    v16bf a1 = frag_lds(sA, mw + 16, lane);
#pragma unroll
    for (int j = 0; j < 4; ++j) {
      v16bf b = frag_lds(sB, nw + 16 * j, lane);
      acc[0][j] = __builtin_amdgcn_wmma_f32_16x16x32_bf16(false, a0, false, b,
                                                          (short)0, acc[0][j], false, false);
      acc[1][j] = __builtin_amdgcn_wmma_f32_16x16x32_bf16(false, a1, false, b,
                                                          (short)0, acc[1][j], false, false);
    }
  }

  // ---- store: VGPR r, lanes 0-15 -> (M=r, N=lane); lanes 16-31 -> (M=r+8) --
  int col     = n0 + nw + (lane & 15);
  int rowbase = m0 + mw + ((lane >> 4) << 3);
#pragma unroll
  for (int i = 0; i < 2; ++i)
#pragma unroll
    for (int j = 0; j < 4; ++j)
#pragma unroll
      for (int r = 0; r < 8; ++r)
        C[(size_t)(rowbase + i * 16 + r) * ldc + col + j * 16] = acc[i][j][r];
}

// ---- kernel 4: depthwise causal conv (k=4) + bias + SiLU -------------------

__global__ __launch_bounds__(256)
void k_conv_silu(const float* __restrict__ xz, const float* __restrict__ cw,
                 const float* __restrict__ cb, float* __restrict__ xc,
                 __bf16* __restrict__ xcb) {
  size_t i = (size_t)blockIdx.x * 256 + threadIdx.x;   // over ROWS*DI
  if (i >= (size_t)ROWS * DI) return;
  int    d  = (int)(i % DI);
  size_t bl = i / DI;
  int    t  = (int)(bl % LL);
  int    b  = (int)(bl / LL);
  float acc = cb[d];
#pragma unroll
  for (int j = 0; j < DC; ++j) {
    int tt = t - (DC - 1) + j;
    if (tt >= 0)
      acc += cw[d * DC + j] * xz[((size_t)b * LL + tt) * (2 * DI) + d];
  }
  float s = siluf(acc);
  xc[bl * DI + d]  = s;
  xcb[bl * DI + d] = f2bf(s);
}

// ---- kernel 5: delta = softplus(dt @ W_dt^T + b_dt)  (K=48, scalar) --------

__global__ __launch_bounds__(256)
void k_delta(const float* __restrict__ proj, const float* __restrict__ Wdt,
             const float* __restrict__ bdt, float* __restrict__ delta) {
  size_t i = (size_t)blockIdx.x * 256 + threadIdx.x;   // over ROWS*DI
  if (i >= (size_t)ROWS * DI) return;
  int    d   = (int)(i % DI);
  size_t row = i / DI;
  const float* pr = proj + row * NPADX;   // dt at cols 0..47
  const float* w  = Wdt + (size_t)d * DTR;
  float acc = bdt[d];
#pragma unroll 8
  for (int r = 0; r < DTR; ++r) acc += pr[r] * w[r];
  delta[row * DI + d] = softplusf(acc);
}

// ---- kernel 6: selective scan + D-skip + SiLU gate -------------------------
// One thread per (b, d) channel; 16 recurrent states in registers.

__global__ __launch_bounds__(256)
void k_scan(const float* __restrict__ delta, const float* __restrict__ xc,
            const float* __restrict__ proj,  const float* __restrict__ A_log,
            const float* __restrict__ Dskip, const float* __restrict__ xz,
            float* __restrict__ y, __bf16* __restrict__ yb) {
  int idx = blockIdx.x * 256 + threadIdx.x;            // 0..BB*DI-1
  if (idx >= BB * DI) return;
  int d = idx % DI;
  int b = idx / DI;

  float A[DS], h[DS];
#pragma unroll
  for (int s = 0; s < DS; ++s) { A[s] = -__expf(A_log[d * DS + s]); h[s] = 0.0f; }
  float Dv = Dskip[d];

  for (int t = 0; t < LL; ++t) {
    size_t row = (size_t)b * LL + t;
    float dl = delta[row * DI + d];
    float xv = xc[row * DI + d];
    const float* pB = proj + row * NPADX + DTR;        // Bm: cols 48..63
    const float* pC = proj + row * NPADX + DTR + DS;   // Cm: cols 64..79
    float dx = dl * xv;
    float yacc = 0.0f;
#pragma unroll
    for (int s = 0; s < DS; ++s) {
      float dA = __expf(dl * A[s]);
      h[s] = dA * h[s] + dx * pB[s];
      yacc = fmaf(h[s], pC[s], yacc);
    }
    float z = xz[row * (2 * DI) + DI + d];             // gate z
    float out = (yacc + Dv * xv) * siluf(z);
    y[row * DI + d]  = out;
    yb[row * DI + d] = f2bf(out);
  }
}

// ---------------------------------------------------------------------------

extern "C" void kernel_launch(void* const* d_in, const int* in_sizes, int n_in,
                              void* d_out, int out_size, void* d_ws, size_t ws_size,
                              hipStream_t stream) {
  const float* x        = (const float*)d_in[0];
  const float* residual = (const float*)d_in[1];
  const float* ln_gamma = (const float*)d_in[2];
  const float* ln_beta  = (const float*)d_in[3];
  const float* W_in     = (const float*)d_in[4];   // [3072, 768]
  const float* conv_w   = (const float*)d_in[5];   // [1536, 4]
  const float* conv_b   = (const float*)d_in[6];
  const float* W_xproj  = (const float*)d_in[7];   // [80, 1536]
  const float* W_dt     = (const float*)d_in[8];   // [1536, 48]
  const float* b_dt     = (const float*)d_in[9];
  const float* A_log    = (const float*)d_in[10];  // [1536, 16]
  const float* D_skip   = (const float*)d_in[11];
  const float* W_out    = (const float*)d_in[12];  // [768, 1536]

  float* out_hidden = (float*)d_out;                       // [B,L,768]
  float* out_resid  = (float*)d_out + (size_t)ROWS * DM;   // [B,L,768]

  // carve workspace (256B aligned)
  char* p = (char*)d_ws;
  auto carve = [&](size_t bytes) -> void* {
    void* r = (void*)p;
    p += (bytes + 255) & ~(size_t)255;
    return r;
  };
  float*  xn      = (float*) carve((size_t)ROWS * DM * 4);
  __bf16* xn_bf   = (__bf16*)carve((size_t)ROWS * DM * 2);
  __bf16* Win_bf  = (__bf16*)carve((size_t)2 * DI * DM * 2);
  float*  xz      = (float*) carve((size_t)ROWS * 2 * DI * 4);
  float*  xc      = (float*) carve((size_t)ROWS * DI * 4);
  __bf16* xc_bf   = (__bf16*)carve((size_t)ROWS * DI * 2);
  __bf16* Wxp_bf  = (__bf16*)carve((size_t)NPADX * DI * 2);
  float*  proj    = (float*) carve((size_t)ROWS * NPADX * 4);
  float*  delta   = (float*) carve((size_t)ROWS * DI * 4);
  float*  yv      = (float*) carve((size_t)ROWS * DI * 4);
  __bf16* y_bf    = (__bf16*)carve((size_t)ROWS * DI * 2);
  __bf16* Wout_bf = (__bf16*)carve((size_t)DM * DI * 2);

  // 1. residual add + layernorm
  k_addres_ln<<<ROWS, 256, 0, stream>>>(x, residual, ln_gamma, ln_beta, out_resid, xn);

  // 2. bf16 conversions
  {
    int n = ROWS * DM;
    k_cvt_bf16<<<(n + 255) / 256, 256, 0, stream>>>(xn, xn_bf, n);
    n = 2 * DI * DM;
    k_cvt_bf16<<<(n + 255) / 256, 256, 0, stream>>>(W_in, Win_bf, n);
    n = NPADX * DI;
    k_cvt_pad_xproj<<<(n + 255) / 256, 256, 0, stream>>>(W_xproj, Wxp_bf);
    n = DM * DI;
    k_cvt_bf16<<<(n + 255) / 256, 256, 0, stream>>>(W_out, Wout_bf, n);
  }

  // 3. in_proj GEMM: xz[4096,3072] = xn[4096,768] * W_in[3072,768]^T
  {
    int blocks = (ROWS / TILE_M) * ((2 * DI) / TILE_N);   // 32 * 24
    k_gemm_nt_bf16<<<blocks, 256, 0, stream>>>(xn_bf, Win_bf, xz,
                                               ROWS, 2 * DI, DM, 2 * DI);
  }

  // 4. depthwise conv + SiLU (+ bf16 copy for x_proj GEMM)
  {
    size_t n = (size_t)ROWS * DI;
    k_conv_silu<<<(int)((n + 255) / 256), 256, 0, stream>>>(xz, conv_w, conv_b, xc, xc_bf);
  }

  // 5. x_proj GEMM: proj[4096,128pad] = xc[4096,1536] * Wxp[128,1536]^T
  {
    int blocks = (ROWS / TILE_M) * (NPADX / TILE_N);      // 32 * 1
    k_gemm_nt_bf16<<<blocks, 256, 0, stream>>>(xc_bf, Wxp_bf, proj,
                                               ROWS, NPADX, DI, NPADX);
  }

  // 6. delta = softplus(dt @ W_dt^T + b_dt)
  {
    size_t n = (size_t)ROWS * DI;
    k_delta<<<(int)((n + 255) / 256), 256, 0, stream>>>(proj, W_dt, b_dt, delta);
  }

  // 7. selective scan + skip + gate
  k_scan<<<(BB * DI + 255) / 256, 256, 0, stream>>>(delta, xc, proj, A_log,
                                                    D_skip, xz, yv, y_bf);

  // 8. out_proj GEMM: hidden[4096,768] = y[4096,1536] * W_out[768,1536]^T
  {
    int blocks = (ROWS / TILE_M) * (DM / TILE_N);         // 32 * 6
    k_gemm_nt_bf16<<<blocks, 256, 0, stream>>>(y_bf, Wout_bf, out_hidden,
                                               ROWS, DM, DI, DM);
  }
}
